// Model_66546223284545
// MI455X (gfx1250) — matrix operand order repaired and checked by reading, not compiled, not run on hardware
//
#include <hip/hip_runtime.h>
#include <hip/hip_bf16.h>
#include <math.h>

#define B_    16
#define T_    384
#define NP_   115
#define NA_   3
#define DIM_  896
#define H_    128
#define DH_   7
#define WIN_  5
#define BT_   (B_*T_)
#define INDIM_ 1140
#define INP_   1152   // padded K for the input GEMM (multiple of 32)

typedef __attribute__((ext_vector_type(16))) _Float16 v16h;
typedef __attribute__((ext_vector_type(8)))  float    v8f;
typedef __attribute__((ext_vector_type(4)))  float    f4;
typedef __attribute__((ext_vector_type(4)))  _Float16 h4;
typedef __attribute__((ext_vector_type(8)))  _Float16 h8;

__device__ __forceinline__ float eluf(float x){ return x > 0.f ? x : __expf(x) - 1.f; }

#define LDS_STRIDE 40   // padded row stride in halves (80B: 16B aligned, bank-conflict free)

// ---------------------------------------------------------------------------
// Fast GEMM: C[M,N] = A[M,K] @ W[K,N] + bias (bias required).
// Requires M%128==0, N%64==0, K%32==0.
// 128x64 tile / 128 threads: 4 waves, each wave owns 2 x 16-row strips and
// 4 x 16-col subtiles = 8 WMMA accumulators.  Double-buffered LDS tiles with
// software pipelining: global b128 prefetch for step k+1 issued between the
// fragment ds_load_b128s and the 8 WMMAs of step k.  B tile stored transposed
// (Bt[N][K]) so each lane's 16 K-halves are one 2x ds_load_b128 fetch.
// ---------------------------------------------------------------------------
__global__ __launch_bounds__(128)
void gemm_fast(const float* __restrict__ A, const float* __restrict__ W,
               const float* __restrict__ bias, float* __restrict__ C,
               int M, int N, int K)
{
  __shared__ _Float16 As[2][128 * LDS_STRIDE];
  __shared__ _Float16 Bt[2][64 * LDS_STRIDE];
  const int tid  = threadIdx.x;
  const int lane = tid & 31;
  const int wave = tid >> 5;
  const int bm = blockIdx.y * 128;
  const int bn = blockIdx.x * 64;
  const int sel = lane >> 4;
  const int rm  = lane & 15;

  // A loader: rows ra+{0,16,...,112}, cols ca..ca+3 of the 128x32 tile
  const int ra = tid >> 3;
  const int ca = (tid & 7) * 4;
  // B loader: rows rb+{0,8,16,24}, cols cb..cb+3 of the 32x64 tile
  const int rb = tid >> 4;
  const int cb = (tid & 15) * 4;

  const float* Ap = A + (size_t)(bm + ra) * K + ca;
  const float* Wp = W + (size_t)rb * N + bn + cb;

  // ---- prologue: stage tile k0=0 into buffer 0 ----
#pragma unroll
  for (int s = 0; s < 8; ++s) {
    f4 av = *(const f4*)(Ap + (size_t)(s * 16) * K);
    h4 ah;
    ah[0] = (_Float16)av[0]; ah[1] = (_Float16)av[1];
    ah[2] = (_Float16)av[2]; ah[3] = (_Float16)av[3];
    *(h4*)&As[0][(ra + s * 16) * LDS_STRIDE + ca] = ah;
  }
#pragma unroll
  for (int s = 0; s < 4; ++s) {
    f4 wv = *(const f4*)(Wp + (size_t)(s * 8) * N);
#pragma unroll
    for (int j = 0; j < 4; ++j)
      Bt[0][(cb + j) * LDS_STRIDE + rb + s * 8] = (_Float16)wv[j];
  }
  __syncthreads();

  v8f acc[2][4] = {};
  int cur = 0;

  for (int k0 = 0; k0 < K; k0 += 32) {
    const int nxt = cur ^ 1;
    const bool more = (k0 + 32) < K;

    // fragment loads from current buffer
    v16h afr[2];
#pragma unroll
    for (int r = 0; r < 2; ++r) {
      const _Float16* ap = &As[cur][(wave * 32 + r * 16 + rm) * LDS_STRIDE + sel * 8];
      h8 x0 = *(const h8*)ap;
      h8 x1 = *(const h8*)(ap + 16);
#pragma unroll
      for (int i = 0; i < 8; ++i) { afr[r][i] = x0[i]; afr[r][i + 8] = x1[i]; }
    }
    v16h bfr[4];
#pragma unroll
    for (int nt = 0; nt < 4; ++nt) {
      const _Float16* bp = &Bt[cur][(nt * 16 + rm) * LDS_STRIDE + sel * 8];
      h8 x0 = *(const h8*)bp;
      h8 x1 = *(const h8*)(bp + 16);
#pragma unroll
      for (int i = 0; i < 8; ++i) { bfr[nt][i] = x0[i]; bfr[nt][i + 8] = x1[i]; }
    }

    // prefetch next tile from global (independent of the WMMAs below)
    f4 av[8], wv[4];
    if (more) {
#pragma unroll
      for (int s = 0; s < 8; ++s)
        av[s] = *(const f4*)(Ap + (size_t)(s * 16) * K + k0 + 32);
#pragma unroll
      for (int s = 0; s < 4; ++s)
        wv[s] = *(const f4*)(Wp + (size_t)(k0 + 32 + s * 8) * N);
    }

    // 8 WMMAs
#pragma unroll
    for (int r = 0; r < 2; ++r)
#pragma unroll
      for (int nt = 0; nt < 4; ++nt)
        acc[r][nt] = __builtin_amdgcn_wmma_f32_16x16x32_f16(
            false, afr[r], false, bfr[nt], (short)0, acc[r][nt], false, false);

    // stage next tile
    if (more) {
#pragma unroll
      for (int s = 0; s < 8; ++s) {
        h4 ah;
        ah[0] = (_Float16)av[s][0]; ah[1] = (_Float16)av[s][1];
        ah[2] = (_Float16)av[s][2]; ah[3] = (_Float16)av[s][3];
        *(h4*)&As[nxt][(ra + s * 16) * LDS_STRIDE + ca] = ah;
      }
#pragma unroll
      for (int s = 0; s < 4; ++s)
#pragma unroll
        for (int j = 0; j < 4; ++j)
          Bt[nxt][(cb + j) * LDS_STRIDE + rb + s * 8] = (_Float16)wv[s][j];
    }
    __syncthreads();
    cur = nxt;
  }

  // epilogue: bias is mandatory on the fast path (no branchy null check)
#pragma unroll
  for (int nt = 0; nt < 4; ++nt) {
    const int col = bn + nt * 16 + rm;
    const float bv = bias[col];
#pragma unroll
    for (int r = 0; r < 2; ++r) {
#pragma unroll
      for (int j = 0; j < 8; ++j) {
        int row = bm + wave * 32 + r * 16 + j + 8 * sel;
        C[(size_t)row * N + col] = acc[r][nt][j] + bv;
      }
    }
  }
}

// ---------------------------------------------------------------------------
// Guarded generic GEMM (small / edge shapes only: M=16 GEMMs, final N=60).
// ---------------------------------------------------------------------------
__global__ __launch_bounds__(128)
void gemm_guard(const float* __restrict__ A, const float* __restrict__ W,
                const float* __restrict__ bias, float* __restrict__ C,
                int M, int N, int K)
{
  __shared__ _Float16 As[64 * LDS_STRIDE];
  __shared__ _Float16 Bt[64 * LDS_STRIDE];
  const int tid  = threadIdx.x;
  const int lane = tid & 31;
  const int wave = tid >> 5;
  const int bm = blockIdx.y * 64;
  const int bn = blockIdx.x * 64;
  const int sel = lane >> 4;
  const int rm  = lane & 15;

  v8f acc[4] = {};

  for (int k0 = 0; k0 < K; k0 += 32) {
    for (int i = tid; i < 64 * 32; i += 128) {
      int r = i >> 5, c = i & 31;
      int gr = bm + r, gc = k0 + c;
      float v = (gr < M && gc < K) ? A[(size_t)gr * K + gc] : 0.f;
      As[r * LDS_STRIDE + c] = (_Float16)v;
    }
    for (int i = tid; i < 32 * 64; i += 128) {
      int r = i >> 6, c = i & 63;
      int gr = k0 + r, gc = bn + c;
      float v = (gr < K && gc < N) ? W[(size_t)gr * N + gc] : 0.f;
      Bt[c * LDS_STRIDE + r] = (_Float16)v;
    }
    __syncthreads();

    const int ar = wave * 16 + rm;
    h8 a0 = *(const h8*)&As[ar * LDS_STRIDE + sel * 8];
    h8 a1 = *(const h8*)&As[ar * LDS_STRIDE + 16 + sel * 8];
    v16h a;
#pragma unroll
    for (int i = 0; i < 8; ++i) { a[i] = a0[i]; a[i + 8] = a1[i]; }
#pragma unroll
    for (int nt = 0; nt < 4; ++nt) {
      const _Float16* bp = &Bt[(nt * 16 + rm) * LDS_STRIDE + sel * 8];
      h8 b0 = *(const h8*)bp;
      h8 b1 = *(const h8*)(bp + 16);
      v16h b;
#pragma unroll
      for (int i = 0; i < 8; ++i) { b[i] = b0[i]; b[i + 8] = b1[i]; }
      acc[nt] = __builtin_amdgcn_wmma_f32_16x16x32_f16(
          false, a, false, b, (short)0, acc[nt], false, false);
    }
    __syncthreads();
  }

#pragma unroll
  for (int nt = 0; nt < 4; ++nt) {
    int col = bn + nt * 16 + rm;
    if (col >= N) continue;
    float bv = bias ? bias[col] : 0.f;
#pragma unroll
    for (int j = 0; j < 8; ++j) {
      int row = bm + wave * 16 + j + 8 * sel;
      if (row < M) C[(size_t)row * N + col] = acc[nt][j] + bv;
    }
  }
}

// ---------------------------------------------------------------------------
// Zero-pad i1w (1140 x 1792) -> (1152 x 1792) staging copy.
// ---------------------------------------------------------------------------
__global__ void pad_w_kernel(const float* __restrict__ src, float* __restrict__ dst)
{
  int idx = blockIdx.x * 256 + threadIdx.x;
  if (idx >= INP_ * 2 * DIM_) return;
  int k = idx / (2 * DIM_);
  dst[idx] = (k < INDIM_) ? src[idx] : 0.f;
}

// ---------------------------------------------------------------------------
// Masked per-(batch,axis) stats over (T,NP) for axis_ln dims=(1,2)
// ---------------------------------------------------------------------------
__global__ __launch_bounds__(256)
void batch_stats(const float* __restrict__ x, float* __restrict__ stats)
{
  int b = blockIdx.x / NA_;
  int a = blockIdx.x % NA_;
  float sw = 0, sx = 0, sxx = 0;
  const int NEl = T_ * NP_;
  for (int i = threadIdx.x; i < NEl; i += 256) {
    int t = i / NP_, p = i % NP_;
    float v = x[(((size_t)b * T_ + t) * NP_ + p) * NA_ + a];
    float w = (v != 0.f) ? 1.f : 0.f;
    sw += w; sx += v * w; sxx += v * v * w;
  }
  __shared__ float r0[256], r1[256], r2[256];
  r0[threadIdx.x] = sw; r1[threadIdx.x] = sx; r2[threadIdx.x] = sxx;
  __syncthreads();
  for (int off = 128; off > 0; off >>= 1) {
    if (threadIdx.x < off) {
      r0[threadIdx.x] += r0[threadIdx.x + off];
      r1[threadIdx.x] += r1[threadIdx.x + off];
      r2[threadIdx.x] += r2[threadIdx.x + off];
    }
    __syncthreads();
  }
  if (threadIdx.x == 0) {
    float denom = r0[0] + 1e-5f;
    float m = r1[0] / denom;
    float var = r2[0] / denom - 2.f * m * (r1[0] / denom) + m * m * (r0[0] / denom);
    stats[(b * NA_ + a) * 2 + 0] = m;
    stats[(b * NA_ + a) * 2 + 1] = sqrtf(fmaxf(var, 0.f));
  }
}

// ---------------------------------------------------------------------------
// Per-token feature builder: nx(345) | nf(345) | lh(225) | rh(225) | pad(12)=0
// Row stride INP_.  Also computes nonmask[b*T+t].
// ---------------------------------------------------------------------------
__global__ __launch_bounds__(128)
void feat_kernel(const float* __restrict__ x,
                 const float* __restrict__ xg, const float* __restrict__ xb,
                 const float* __restrict__ fg0, const float* __restrict__ fb0,
                 const float* __restrict__ fg1, const float* __restrict__ fb1,
                 const float* __restrict__ fg2, const float* __restrict__ fb2,
                 const float* __restrict__ fg3, const float* __restrict__ fb3,
                 const float* __restrict__ stats,
                 float* __restrict__ feat, float* __restrict__ nonmask)
{
  const int bt = blockIdx.x;
  const int b  = bt / T_;
  __shared__ float xs[NP_ * NA_];
  __shared__ float nxs[NP_ * NA_];
  __shared__ float gm[12], gs[12];
  __shared__ int s_any;
  if (threadIdx.x == 0) s_any = 0;
  for (int i = threadIdx.x; i < NP_ * NA_; i += 128)
    xs[i] = x[(size_t)bt * NP_ * NA_ + i];
  __syncthreads();

  int f = 0;
  for (int i = threadIdx.x; i < NP_ * NA_; i += 128) f |= (xs[i] != 0.f);
  if (f) atomicOr(&s_any, 1);

  if (threadIdx.x < 12) {
    const int rs_[4] = {0, 33, 73, 94};
    const int re_[4] = {33, 73, 94, 115};
    int r = threadIdx.x / 3, a = threadIdx.x % 3;
    float sw = 0, sx = 0, sxx = 0;
    for (int p = rs_[r]; p < re_[r]; ++p) {
      float v = xs[p * 3 + a];
      float w = (v != 0.f) ? 1.f : 0.f;
      sw += w; sx += v * w; sxx += v * v * w;
    }
    float denom = sw + 1e-5f;
    float m = sx / denom;
    float var = sxx / denom - 2.f * m * (sx / denom) + m * m * (sw / denom);
    gm[threadIdx.x] = m;
    gs[threadIdx.x] = sqrtf(fmaxf(var, 0.f));
  }
  __syncthreads();
  if (threadIdx.x == 0) nonmask[bt] = s_any ? 1.f : 0.f;

  float* fo = feat + (size_t)bt * INP_;
  for (int i = threadIdx.x; i < NP_ * NA_; i += 128) {
    int a = i % 3;
    float m = stats[(b * 3 + a) * 2], sd = stats[(b * 3 + a) * 2 + 1];
    float v = xs[i];
    float w = (v != 0.f) ? 1.f : 0.f;
    float val = (xg[i] * ((v - m) / (sd + 1e-5f)) + xb[i]) * w;
    nxs[i] = val;
    fo[i] = val;
  }
  __syncthreads();
  for (int i = threadIdx.x; i < NP_ * NA_; i += 128) {
    int p = i / 3, a = i % 3;
    int r = (p < 33) ? 0 : (p < 73) ? 1 : (p < 94) ? 2 : 3;
    const int rs_[4] = {0, 33, 73, 94};
    const float* fgp = (r == 0) ? fg0 : (r == 1) ? fg1 : (r == 2) ? fg2 : fg3;
    const float* fbp = (r == 0) ? fb0 : (r == 1) ? fb1 : (r == 2) ? fb2 : fb3;
    int pl = p - rs_[r];
    float v = xs[i];
    float w = (v != 0.f) ? 1.f : 0.f;
    float m = gm[r * 3 + a], sd = gs[r * 3 + a];
    fo[NP_ * NA_ + i] = (fgp[pl * 3 + a] * ((v - m) / (sd + 1e-5f)) + fbp[pl * 3 + a]) * w;
  }
  for (int j = threadIdx.x; j < 2 * 225; j += 128) {
    int hand = j / 225, lc = j % 225;
    int base = (hand == 0) ? 73 : 94;
    float o;
    if (lc < 210) {
      int rr = 0, cum = 0;
      while (cum + (20 - rr) <= lc) { cum += 20 - rr; ++rr; }
      int cc = lc - cum;
      float dx = nxs[(base + rr) * 3 + 0] - nxs[(base + cc) * 3 + 0];
      float dy = nxs[(base + rr) * 3 + 1] - nxs[(base + cc) * 3 + 1];
      o = sqrtf(dx * dx + dy * dy);
    } else {
      int k = lc - 210;
      int fi = k / 3, ii = k % 3 + 1;
      int i0 = (ii - 1 == 0) ? 0 : fi * 4 + (ii - 1);
      int i1 = fi * 4 + ii;
      int i2 = fi * 4 + (ii + 1);
      float ax = nxs[(base + i0) * 3 + 0] - nxs[(base + i1) * 3 + 0];
      float ay = nxs[(base + i0) * 3 + 1] - nxs[(base + i1) * 3 + 1];
      float bx = nxs[(base + i2) * 3 + 0] - nxs[(base + i1) * 3 + 0];
      float by = nxs[(base + i2) * 3 + 1] - nxs[(base + i1) * 3 + 1];
      float na = fmaxf(sqrtf(ax * ax + ay * ay), 1e-5f);
      float nb = fmaxf(sqrtf(bx * bx + by * by), 1e-5f);
      o = (ax * bx + ay * by) / (na * nb);
    }
    fo[2 * NP_ * NA_ + j] = o;
  }
  if (threadIdx.x < INP_ - INDIM_) fo[INDIM_ + threadIdx.x] = 0.f;  // K pad
}

// ---------------------------------------------------------------------------
// Row LayerNorm (+ optional ELU, + optional residual). One block per row.
// ---------------------------------------------------------------------------
__global__ __launch_bounds__(256)
void ln_act(const float* __restrict__ in, float* __restrict__ out,
            const float* __restrict__ g, const float* __restrict__ b,
            const float* __restrict__ res, int N, int do_elu)
{
  size_t row = blockIdx.x;
  const float* ip = in + row * (size_t)N;
  float s1 = 0, s2 = 0;
  for (int i = threadIdx.x; i < N; i += 256) { float v = ip[i]; s1 += v; s2 += v * v; }
  __shared__ float r1[256], r2[256];
  r1[threadIdx.x] = s1; r2[threadIdx.x] = s2;
  __syncthreads();
  for (int off = 128; off > 0; off >>= 1) {
    if (threadIdx.x < off) {
      r1[threadIdx.x] += r1[threadIdx.x + off];
      r2[threadIdx.x] += r2[threadIdx.x + off];
    }
    __syncthreads();
  }
  float m = r1[0] / N;
  float var = r2[0] / N - m * m;
  float rs = 1.f / sqrtf(var + 1e-5f);
  float* op = out + row * (size_t)N;
  const float* rp = res ? res + row * (size_t)N : nullptr;
  for (int i = threadIdx.x; i < N; i += 256) {
    float v = (ip[i] - m) * rs * g[i] + b[i];
    if (do_elu) v = eluf(v);
    if (rp) v += rp[i];
    op[i] = v;
  }
}

// ---------------------------------------------------------------------------
// Elementwise kernels
// ---------------------------------------------------------------------------
__global__ void exp_mask(float* __restrict__ a, const float* __restrict__ nm)
{
  int idx = blockIdx.x * 256 + threadIdx.x;
  if (idx >= BT_ * H_) return;
  a[idx] = __expf(a[idx]) * nm[idx / H_];
}

__global__ void exp_k(float* __restrict__ a, int n)
{
  int idx = blockIdx.x * 256 + threadIdx.x;
  if (idx < n) a[idx] = __expf(a[idx]);
}

__global__ void pos_add(float* __restrict__ h)
{
  int idx = blockIdx.x * 256 + threadIdx.x;
  if (idx >= BT_ * DIM_) return;
  int d = idx % DIM_;
  int t = (idx / DIM_) % T_;
  int k2 = (d >> 1) * 2;
  float freq = __expf(-(float)k2 * 9.210340372f / (float)DIM_);  // ln(10000)
  float ang = (float)t * freq;
  h[idx] += (d & 1) ? __cosf(ang) : __sinf(ang);
}

// ---------------------------------------------------------------------------
// g[b,h,dh] = sum_t attn[b,t,h] * v[b,t,h,dh], attn = aexp / (sum_t aexp + 1e-5)
// ---------------------------------------------------------------------------
__global__ __launch_bounds__(128)
void gv_kernel(const float* __restrict__ aexp, const float* __restrict__ v,
               float* __restrict__ g)
{
  int b = blockIdx.x / H_;
  int h = blockIdx.x % H_;
  float s = 0.f, p[DH_];
#pragma unroll
  for (int i = 0; i < DH_; ++i) p[i] = 0.f;
  for (int t = threadIdx.x; t < T_; t += 128) {
    float a = aexp[((size_t)b * T_ + t) * H_ + h];
    s += a;
    const float* vp = v + ((size_t)b * T_ + t) * DIM_ + h * DH_;
#pragma unroll
    for (int i = 0; i < DH_; ++i) p[i] += a * vp[i];
  }
  __shared__ float red[8][128];
  red[0][threadIdx.x] = s;
#pragma unroll
  for (int i = 0; i < DH_; ++i) red[1 + i][threadIdx.x] = p[i];
  __syncthreads();
  for (int off = 64; off > 0; off >>= 1) {
    if (threadIdx.x < off)
      for (int i = 0; i < 8; ++i) red[i][threadIdx.x] += red[i][threadIdx.x + off];
    __syncthreads();
  }
  if (threadIdx.x < DH_) {
    float inv = 1.f / (red[0][0] + 1e-5f);
    g[((size_t)b * H_ + h) * DH_ + threadIdx.x] = red[1 + threadIdx.x][0] * inv;
  }
}

// ---------------------------------------------------------------------------
// Windowed attention recombine (replicates reference pairing exactly).
// ---------------------------------------------------------------------------
__global__ __launch_bounds__(256)
void window_kernel(const float* __restrict__ aexp, const float* __restrict__ v,
                   const float* __restrict__ g, const float* __restrict__ pos,
                   float* __restrict__ out, int dil)
{
  int idx = blockIdx.x * 256 + threadIdx.x;
  if (idx >= BT_ * H_) return;
  int h  = idx % H_;
  int bt = idx / H_;
  int b = bt / T_, t = bt % T_;
  const float* pp = pos + (size_t)b * (WIN_ + 1) * H_;
  float e[6];
  e[0] = pp[h];
  float s = e[0];
#pragma unroll
  for (int j = 1; j <= 5; ++j) {
    int t2 = t + dil * (j - 3);
    float a = (t2 >= 0 && t2 < T_) ? aexp[((size_t)b * T_ + t2) * H_ + h] : 0.f;
    e[j] = a * pp[j * H_ + h];
    s += e[j];
  }
  s += 1e-5f;
  float c[5];
#pragma unroll
  for (int j = 0; j < 5; ++j) c[j] = e[j] / s;

  const float* gp = g + ((size_t)b * H_ + h) * DH_;
  float acc[DH_];
#pragma unroll
  for (int dh = 0; dh < DH_; ++dh) acc[dh] = c[0] * gp[dh];
#pragma unroll
  for (int j = 0; j < 5; ++j) {
    int t2 = t + dil * (j - 2);
    if (t2 >= 0 && t2 < T_) {
      const float* vp = v + ((size_t)b * T_ + t2) * DIM_ + h * DH_;
#pragma unroll
      for (int dh = 0; dh < DH_; ++dh) acc[dh] += c[j] * vp[dh];
    }
  }
  float* op = out + (size_t)bt * DIM_ + h * DH_;
#pragma unroll
  for (int dh = 0; dh < DH_; ++dh) op[dh] = acc[dh];
}

// ---------------------------------------------------------------------------
// Orchestration
// ---------------------------------------------------------------------------
extern "C" void kernel_launch(void* const* d_in, const int* in_sizes, int n_in,
                              void* d_out, int out_size, void* d_ws, size_t ws_size,
                              hipStream_t stream)
{
  (void)in_sizes; (void)n_in; (void)out_size; (void)ws_size;
  const float* x  = (const float*)d_in[0];
  const float* xg = (const float*)d_in[1];
  const float* xb = (const float*)d_in[2];
  const float* fg[4]; const float* fb[4];
  for (int r = 0; r < 4; ++r) { fg[r] = (const float*)d_in[3 + 2 * r]; fb[r] = (const float*)d_in[4 + 2 * r]; }
  const float* i1w  = (const float*)d_in[11];
  const float* i1b  = (const float*)d_in[12];
  const float* i1lg = (const float*)d_in[13];
  const float* i1lb = (const float*)d_in[14];
  const float* i2w  = (const float*)d_in[15];
  const float* i2b  = (const float*)d_in[16];
  const float* i2lg = (const float*)d_in[17];
  const float* i2lb = (const float*)d_in[18];
  auto LP = [&](int l, int k) { return (const float*)d_in[19 + 16 * l + k]; };
  const float* fow = (const float*)d_in[147];
  const float* fob = (const float*)d_in[148];

  float* w = (float*)d_ws;
  size_t o = 0;
  auto alloc = [&](size_t n) { float* p = w + o; o += n; return p; };
  float* feat = alloc((size_t)BT_ * INP_);
  float* i1wp = alloc((size_t)INP_ * 2 * DIM_);
  float* big  = alloc((size_t)BT_ * 2 * DIM_);
  float* hA   = alloc((size_t)BT_ * DIM_);
  float* hB   = alloc((size_t)BT_ * DIM_);
  float* vbuf = alloc((size_t)BT_ * DIM_);
  float* outt = alloc((size_t)BT_ * DIM_);
  float* aexp = alloc((size_t)BT_ * H_);
  float* gbuf = alloc((size_t)B_ * DIM_);
  float* h1b  = alloc((size_t)B_ * H_);
  float* posb = alloc((size_t)B_ * (WIN_ + 1) * H_);
  float* nonm = alloc((size_t)BT_);
  float* stats = alloc(96);

  dim3 blk128(128), blk256(256);
  auto gemm = [&](const float* A, const float* W_, const float* bias, float* C,
                  int M, int N, int K) {
    if ((M % 128) == 0 && (N % 64) == 0 && (K % 32) == 0 && bias) {
      dim3 grid(N / 64, M / 128);
      gemm_fast<<<grid, blk128, 0, stream>>>(A, W_, bias, C, M, N, K);
    } else {
      dim3 grid((N + 63) / 64, (M + 63) / 64);
      gemm_guard<<<grid, blk128, 0, stream>>>(A, W_, bias, C, M, N, K);
    }
  };

  batch_stats<<<dim3(B_ * NA_), blk256, 0, stream>>>(x, stats);
  feat_kernel<<<dim3(BT_), blk128, 0, stream>>>(x, xg, xb,
      fg[0], fb[0], fg[1], fb[1], fg[2], fb[2], fg[3], fb[3],
      stats, feat, nonm);
  pad_w_kernel<<<dim3((INP_ * 2 * DIM_ + 255) / 256), blk256, 0, stream>>>(i1w, i1wp);

  gemm(feat, i1wp, i1b, big, BT_, 2 * DIM_, INP_);
  ln_act<<<dim3(BT_), blk256, 0, stream>>>(big, big, i1lg, i1lb, nullptr, 2 * DIM_, 1);
  gemm(big, i2w, i2b, hA, BT_, DIM_, 2 * DIM_);
  ln_act<<<dim3(BT_), blk256, 0, stream>>>(hA, hA, i2lg, i2lb, nullptr, DIM_, 1);
  pos_add<<<dim3((BT_ * DIM_ + 255) / 256), blk256, 0, stream>>>(hA);

  float* hc = hA;
  float* hn = hB;
  for (int l = 0; l < 8; ++l) {
    int d = (l == 0) ? 1 : 3;
    const float *aw_ = LP(l, 0),  *ab_ = LP(l, 1),
                *p1w = LP(l, 2),  *p1b = LP(l, 3),
                *plg = LP(l, 4),  *plb = LP(l, 5),
                *p2w = LP(l, 6),  *p2b = LP(l, 7),
                *vw_ = LP(l, 8),  *vb_ = LP(l, 9),
                *vlg = LP(l, 10), *vlb = LP(l, 11),
                *ow_ = LP(l, 12), *ob_ = LP(l, 13),
                *olg = LP(l, 14), *olb = LP(l, 15);

    gemm(hc, aw_, ab_, aexp, BT_, H_, DIM_);
    exp_mask<<<dim3((BT_ * H_ + 255) / 256), blk256, 0, stream>>>(aexp, nonm);

    gemm(hc, vw_, vb_, vbuf, BT_, DIM_, DIM_);
    ln_act<<<dim3(BT_), blk256, 0, stream>>>(vbuf, vbuf, vlg, vlb, nullptr, DIM_, 1);

    gv_kernel<<<dim3(B_ * H_), blk128, 0, stream>>>(aexp, vbuf, gbuf);

    gemm(gbuf, p1w, p1b, h1b, B_, H_, DIM_);
    ln_act<<<dim3(B_), blk256, 0, stream>>>(h1b, h1b, plg, plb, nullptr, H_, 1);
    gemm(h1b, p2w, p2b, posb, B_, (WIN_ + 1) * H_, H_);
    exp_k<<<dim3((B_ * (WIN_ + 1) * H_ + 255) / 256), blk256, 0, stream>>>(
        posb, B_ * (WIN_ + 1) * H_);

    window_kernel<<<dim3((BT_ * H_ + 255) / 256), blk256, 0, stream>>>(
        aexp, vbuf, gbuf, posb, outt, d);

    gemm(outt, ow_, ob_, vbuf, BT_, DIM_, DIM_);
    ln_act<<<dim3(BT_), blk256, 0, stream>>>(vbuf, hn, olg, olb, hc, DIM_, 1);

    float* tmp = hc; hc = hn; hn = tmp;
  }

  gemm(hc, fow, fob, (float*)d_out, BT_, 60, DIM_);
}
